// GNN_Layers_3161095930495
// MI455X (gfx1250) — compile-verified, matrix operand order checked by reading, not run on hardware
//
#include <hip/hip_runtime.h>
#include <hip/hip_bf16.h>

#define NN 100000
#define NE 1600000
#define FIN 256
#define FHID 128
#define BN_EPS 1e-3f

typedef unsigned short ushort_t;
typedef __attribute__((ext_vector_type(16))) __bf16 v16bf;
typedef __attribute__((ext_vector_type(8)))  float  v8f;

union FragBF { v16bf v; unsigned u[8]; uint4 q[2]; };
union FragF  { v8f   v; float    f[8]; };

__device__ __forceinline__ unsigned pack2_bf16(float x, float y) {
    // round-to-nearest-even f32 -> bf16, packed low=x, high=y
    unsigned bx = __float_as_uint(x);
    bx = (bx + 0x7fffu + ((bx >> 16) & 1u)) >> 16;
    unsigned by = __float_as_uint(y);
    by = (by + 0x7fffu + ((by >> 16) & 1u)) >> 16;
    return (bx & 0xffffu) | (by << 16);
}

__device__ __forceinline__ void atomic_add_f32(float* p, float v) {
    __hip_atomic_fetch_add(p, v, __ATOMIC_RELAXED, __HIP_MEMORY_SCOPE_AGENT);
}

// ---------------- elementwise / setup kernels ----------------

__global__ void zero_f32_kernel(float* __restrict__ p, int n) {
    int i = blockIdx.x * blockDim.x + threadIdx.x;
    if (i < n) p[i] = 0.0f;
}

__global__ void deg_accum_kernel(const int* __restrict__ dst,
                                 const float* __restrict__ ew,
                                 float* __restrict__ deg) {
    int e = blockIdx.x * blockDim.x + threadIdx.x;
    if (e < NE) atomic_add_f32(&deg[dst[e]], ew[e]);
}

__global__ void dinv_kernel(float* __restrict__ deg_dinv) {
    int i = blockIdx.x * blockDim.x + threadIdx.x;
    if (i < NN) {
        float d = deg_dinv[i] + 1.0f;   // + self-loop; always > 0
        deg_dinv[i] = rsqrtf(d);
    }
}

// pack fp32 pairs -> bf16 dwords; n2 = total_elems/2
__global__ void f32_to_bf16_kernel(const float* __restrict__ src,
                                   unsigned* __restrict__ dst, int n2) {
    int i = blockIdx.x * blockDim.x + threadIdx.x;
    if (i >= n2) return;
    const float2 v = *(const float2*)(src + (size_t)i * 2);
    dst[i] = pack2_bf16(v.x, v.y);
}

// Wt[n][k] = bf16(W[k][n]); W is [K x Nout] row-major
__global__ void w_transpose_kernel(const float* __restrict__ W,
                                   ushort_t* __restrict__ Wt,
                                   int K, int Nout) {
    int idx = blockIdx.x * blockDim.x + threadIdx.x;
    if (idx >= K * Nout) return;
    int n = idx / K;
    int k = idx - n * K;
    unsigned b = __float_as_uint(W[(size_t)k * Nout + n]);
    b = (b + 0x7fffu + ((b >> 16) & 1u)) >> 16;
    Wt[idx] = (ushort_t)b;
}

// ---------------- WMMA GEMM: C[M x FHID] = A[M x K] * W[K x FHID] ----------------
// A bf16 row-major [M x K]; Wt bf16 stored [FHID x K] (i.e. W^T row-major).
// grid.x = M/16 blocks, 256 threads = 8 waves; wave w -> column tile w (FHID==128).
// Fragment layout (16-bit A 16x32, ISA 7.12.2): lane L holds row M=L%16;
// VGPR r holds K-pair at (r>>2)*16 + (L>>4)*8 + (r&3)*2  -> u[0..3] and u[4..7]
// are each 16 contiguous bytes => two b128 loads per fragment.

template <int K>
__global__ void gemm_wmma_kernel(const ushort_t* __restrict__ A,
                                 const ushort_t* __restrict__ Wt,
                                 float* __restrict__ C) {
    const int lane  = threadIdx.x & 31;
    const int wave  = threadIdx.x >> 5;
    const int mBase = blockIdx.x << 4;
    const int nBase = wave << 4;

    const int l15 = lane & 15;
    const int hi8 = (lane >> 4) << 3;       // 0 or 8

    const ushort_t* arow = A  + (size_t)(mBase + l15) * K + hi8;
    const ushort_t* wrow = Wt + (size_t)(nBase + l15) * K + hi8;

    FragF acc;
#pragma unroll
    for (int r = 0; r < 8; ++r) acc.f[r] = 0.0f;

#pragma unroll
    for (int kk = 0; kk < K; kk += 32) {
        FragBF a, b;
        a.q[0] = *(const uint4*)(arow + kk);        // K = kk+hi8 + 0..7
        a.q[1] = *(const uint4*)(arow + kk + 16);   // K = kk+hi8 + 16..23
        b.q[0] = *(const uint4*)(wrow + kk);
        b.q[1] = *(const uint4*)(wrow + kk + 16);
        acc.v = __builtin_amdgcn_wmma_f32_16x16x32_bf16(
            false, a.v, false, b.v, (short)0, acc.v, false, false);
    }

    // C/D layout: lane holds column n = lane%16; VGPR r holds row (lane/16)*8 + r
    const int n    = nBase + l15;
    const int mtop = (lane >> 4) << 3;
#pragma unroll
    for (int r = 0; r < 8; ++r) {
        C[(size_t)(mBase + mtop + r) * FHID + n] = acc.f[r];
    }
}

// ---------------- GCN aggregate ----------------

// agg[i][f] = dinv[i]^2 * h[i][f] + bias[f]  (self-loop + bias)
__global__ void agg_init_kernel(const float* __restrict__ h,
                                const float* __restrict__ dinv,
                                const float* __restrict__ bias,
                                float* __restrict__ agg) {
    int idx = blockIdx.x * blockDim.x + threadIdx.x;
    if (idx >= NN * FHID) return;
    int i = idx >> 7;
    int f = idx & 127;
    float di = dinv[i];
    agg[idx] = di * di * h[idx] + bias[f];
}

// one wave per edge; each lane handles 4 features (float4)
__global__ void edge_scatter_kernel(const int* __restrict__ src,
                                    const int* __restrict__ dst,
                                    const float* __restrict__ ew,
                                    const float* __restrict__ dinv,
                                    const float* __restrict__ h,
                                    float* __restrict__ agg) {
    int lane = threadIdx.x & 31;
    int wave = threadIdx.x >> 5;
    long long e = (long long)blockIdx.x * 8 + wave;
    if (e >= NE) return;
    int s = src[e];
    int d = dst[e];
    float coef = dinv[s] * ew[e] * dinv[d];
    const float4 hv = *(const float4*)(h + (size_t)s * FHID + lane * 4);
    float* ap = agg + (size_t)d * FHID + lane * 4;
    atomic_add_f32(ap + 0, hv.x * coef);
    atomic_add_f32(ap + 1, hv.y * coef);
    atomic_add_f32(ap + 2, hv.z * coef);
    atomic_add_f32(ap + 3, hv.w * coef);
}

// ---------------- BatchNorm ----------------

__global__ void bn_stats_kernel(const float* __restrict__ buf,
                                float* __restrict__ mean,
                                float* __restrict__ rstd) {
    __shared__ float s_sum[256];
    __shared__ float s_sq[256];
    const int f = blockIdx.x;          // one block per feature
    const int t = threadIdx.x;
    float sum = 0.0f, sq = 0.0f;
    for (int row = t; row < NN; row += 256) {
        float v = buf[(size_t)row * FHID + f];
        sum += v;
        sq  += v * v;
    }
    s_sum[t] = sum; s_sq[t] = sq;
    __syncthreads();
    for (int off = 128; off > 0; off >>= 1) {
        if (t < off) { s_sum[t] += s_sum[t + off]; s_sq[t] += s_sq[t + off]; }
        __syncthreads();
    }
    if (t == 0) {
        float m   = s_sum[0] * (1.0f / NN);
        float var = s_sq[0] * (1.0f / NN) - m * m;   // biased variance
        mean[f] = m;
        rstd[f] = rsqrtf(var + BN_EPS);
    }
}

// normalize + ReLU, fp32 output (final layer -> d_out)
__global__ void bn_norm_relu_f32_kernel(const float* __restrict__ buf,
                                        const float* __restrict__ mean,
                                        const float* __restrict__ rstd,
                                        const float* __restrict__ gamma,
                                        const float* __restrict__ beta,
                                        float* __restrict__ out) {
    int idx = blockIdx.x * blockDim.x + threadIdx.x;
    if (idx >= NN * FHID) return;
    int f = idx & 127;
    float v = (buf[idx] - mean[f]) * rstd[f] * gamma[f] + beta[f];
    out[idx] = v > 0.0f ? v : 0.0f;
}

// normalize + ReLU, packed bf16 output (feeds next layer's WMMA A operand)
__global__ void bn_norm_relu_bf16_kernel(const float* __restrict__ buf,
                                         const float* __restrict__ mean,
                                         const float* __restrict__ rstd,
                                         const float* __restrict__ gamma,
                                         const float* __restrict__ beta,
                                         unsigned* __restrict__ out) {
    int idx = blockIdx.x * blockDim.x + threadIdx.x;  // pair index
    if (idx >= (NN * FHID) / 2) return;
    int e0 = idx * 2;
    int f0 = e0 & 127;
    const float2 v = *(const float2*)(buf + (size_t)e0);
    float y0 = (v.x - mean[f0]) * rstd[f0] * gamma[f0] + beta[f0];
    float y1 = (v.y - mean[f0 + 1]) * rstd[f0 + 1] * gamma[f0 + 1] + beta[f0 + 1];
    y0 = y0 > 0.0f ? y0 : 0.0f;
    y1 = y1 > 0.0f ? y1 : 0.0f;
    out[idx] = pack2_bf16(y0, y1);
}

// ---------------- launch ----------------

extern "C" void kernel_launch(void* const* d_in, const int* in_sizes, int n_in,
                              void* d_out, int out_size, void* d_ws, size_t ws_size,
                              hipStream_t stream) {
    const float* x   = (const float*)d_in[0];
    const int*   ei  = (const int*)d_in[1];      // [2, NE]
    const float* ew  = (const float*)d_in[2];
    const float* W0  = (const float*)d_in[3];
    const float* b0  = (const float*)d_in[4];
    const float* g0  = (const float*)d_in[5];
    const float* be0 = (const float*)d_in[6];
    const float* W1  = (const float*)d_in[7];
    const float* b1  = (const float*)d_in[8];
    const float* g1  = (const float*)d_in[9];
    const float* be1 = (const float*)d_in[10];
    float* out = (float*)d_out;

    const int* src = ei;
    const int* dst = ei + NE;

    // workspace layout
    float* bufA = (float*)d_ws;                       // NN*FHID fp32
    float* bufB = bufA + (size_t)NN * FHID;           // NN*FHID fp32
    float* dinv = bufB + (size_t)NN * FHID;           // NN (deg, then deg^-1/2)
    float* mean = dinv + NN;                          // FHID
    float* rstd = mean + FHID;                        // FHID
    ushort_t* W0t  = (ushort_t*)(rstd + FHID);        // FHID*FIN bf16 (W0^T)
    ushort_t* W1t  = W0t + (size_t)FHID * FIN;        // FHID*FHID bf16 (W1^T)
    ushort_t* xbf  = W1t + (size_t)FHID * FHID;       // NN*FIN bf16
    ushort_t* hnbf = xbf + (size_t)NN * FIN;          // NN*FHID bf16

    const int ELEMS   = NN * FHID;          // 12.8M
    const int EW_GRID = ELEMS / 256;        // 50000
    const int M_TILES = NN / 16;            // 6250
    const int E_GRID  = (NE + 7) / 8;       // 200000 (1 wave per edge, 8 waves/block)

    // degree -> dinv
    zero_f32_kernel<<<(NN + 255) / 256, 256, 0, stream>>>(dinv, NN);
    deg_accum_kernel<<<(NE + 255) / 256, 256, 0, stream>>>(dst, ew, dinv);
    dinv_kernel<<<(NN + 255) / 256, 256, 0, stream>>>(dinv);

    // weights -> bf16 transposed; x -> bf16
    w_transpose_kernel<<<(FIN * FHID + 255) / 256, 256, 0, stream>>>(W0, W0t, FIN, FHID);
    w_transpose_kernel<<<(FHID * FHID + 255) / 256, 256, 0, stream>>>(W1, W1t, FHID, FHID);
    f32_to_bf16_kernel<<<(NN * FIN / 2 + 255) / 256, 256, 0, stream>>>(
        x, (unsigned*)xbf, NN * FIN / 2);

    // ---- layer 0 ----
    gemm_wmma_kernel<FIN><<<M_TILES, 256, 0, stream>>>(xbf, W0t, bufA);        // h0 -> bufA
    agg_init_kernel<<<EW_GRID, 256, 0, stream>>>(bufA, dinv, b0, bufB);        // agg0 -> bufB
    edge_scatter_kernel<<<E_GRID, 256, 0, stream>>>(src, dst, ew, dinv, bufA, bufB);
    bn_stats_kernel<<<FHID, 256, 0, stream>>>(bufB, mean, rstd);
    bn_norm_relu_bf16_kernel<<<EW_GRID / 2, 256, 0, stream>>>(
        bufB, mean, rstd, g0, be0, (unsigned*)hnbf);                            // hn0 (bf16)

    // ---- layer 1 ----
    gemm_wmma_kernel<FHID><<<M_TILES, 256, 0, stream>>>(hnbf, W1t, bufB);      // h1 -> bufB
    agg_init_kernel<<<EW_GRID, 256, 0, stream>>>(bufB, dinv, b1, bufA);        // agg1 -> bufA
    edge_scatter_kernel<<<E_GRID, 256, 0, stream>>>(src, dst, ew, dinv, bufB, bufA);
    bn_stats_kernel<<<FHID, 256, 0, stream>>>(bufA, mean, rstd);
    bn_norm_relu_f32_kernel<<<EW_GRID, 256, 0, stream>>>(bufA, mean, rstd, g1, be1, out);
}
